// AttentionEncoder_47622597378287
// MI455X (gfx1250) — matrix operand order repaired and checked by reading, not compiled, not run on hardware
//
#include <hip/hip_runtime.h>

// ---------------------------------------------------------------------------
// CDNA5 WMMA (wave32): v_wmma_f32_16x16x32_bf16  +  async global->LDS staging
// ---------------------------------------------------------------------------
typedef __attribute__((ext_vector_type(16))) __bf16 v16bf;
typedef __attribute__((ext_vector_type(8)))  float  v8f;

#define HW_  4096
#define C_   512
#define C8_  64

// Hardware bf16 convert (v_cvt_*bf16_f32), RNE.
__device__ __forceinline__ unsigned short f2bf(float f) {
    __bf16 h = (__bf16)f;
    return __builtin_bit_cast(unsigned short, h);
}
// Pack two floats into two bf16 (pairs into v_cvt_pk_bf16_f32 when available).
__device__ __forceinline__ unsigned pack2bf(float x, float y) {
    return (unsigned)f2bf(x) | ((unsigned)f2bf(y) << 16);
}

struct U128x2 { uint4 lo, hi; };
// Build a 16-element bf16 fragment from two contiguous 16-byte LDS chunks.
__device__ __forceinline__ v16bf frag2(const unsigned short* p0, const unsigned short* p1) {
    U128x2 t;
    t.lo = *(const uint4*)p0;
    t.hi = *(const uint4*)p1;
    return __builtin_bit_cast(v16bf, t);
}
// A-matrix fragment (16-bit, M=lo, K-chunk k2): halves at K = k2*32+hi*8 and +16.
__device__ __forceinline__ v16bf fragA(const unsigned short* row, int hi, int k2) {
    return frag2(row + k2 * 32 + hi * 8, row + k2 * 32 + 16 + hi * 8);
}
// B-matrix fragment from Bt[N][K] row: halves at K = k2*32+hi*8 and +16.
__device__ __forceinline__ v16bf fragB(const unsigned short* row, int hi, int k2) {
    return frag2(row + k2 * 32 + hi * 8, row + k2 * 32 + 16 + hi * 8);
}

// GLOBAL_LOAD_ASYNC_TO_LDS_B128: per-lane 16B global -> LDS, tracked by ASYNCcnt.
__device__ __forceinline__ void async_b128(void* lds, const void* gaddr) {
    unsigned lds_addr = (unsigned)(unsigned long long)lds;   // low 32b = LDS address
    asm volatile("global_load_async_to_lds_b128 %0, %1, off"
                 :: "v"(lds_addr), "v"(gaddr) : "memory");
}
__device__ __forceinline__ void wait_async0() {
    asm volatile("s_wait_asynccnt 0x0" ::: "memory");
}

// ---------------------------------------------------------------------------
// Kernel 1: fused QKV 1x1-conv projections as WMMA bf16 GEMMs.
//   blockIdx.x : n-block (64 pixels), blockIdx.y : 0=q,1=k(+pos),2..9=v slice,
//   blockIdx.z : batch.  256 threads = 8 waves, 2 output tiles per wave.
// Workspace (bf16 bits):
//   qWS [B][n][64]   kWS [B][m][64] (pos folded)   vWS [B][C][m]
// ---------------------------------------------------------------------------
__global__ __launch_bounds__(256) void attn_proj_kernel(
    const float* __restrict__ fe,  const float* __restrict__ total,
    const float* __restrict__ Wq,  const float* __restrict__ bq,
    const float* __restrict__ Wk,  const float* __restrict__ bk,
    const float* __restrict__ Wv,  const float* __restrict__ bv,
    const float* __restrict__ ht,  const float* __restrict__ wt,
    unsigned short* __restrict__ qWS, unsigned short* __restrict__ kWS,
    unsigned short* __restrict__ vWS)
{
    const int nb = blockIdx.x;
    const int pj = blockIdx.y;
    const int b  = blockIdx.z;
    const int n0 = nb * 64;

    const float* X; const float* W; const float* bias; int orow0;
    if (pj == 0)      { X = total; W = Wq;                          bias = bq;               orow0 = 0; }
    else if (pj == 1) { X = fe;    W = Wk;                          bias = bk;               orow0 = 0; }
    else              { X = fe;    W = Wv + (size_t)(pj - 2)*64*C_; bias = bv + (pj - 2)*64; orow0 = (pj - 2)*64; }

    __shared__ __align__(16) unsigned short xs[64][40];   // x^T  [n][k], pitch 80B
    __shared__ __align__(16) unsigned short wsm[64][40];  // W    [o][k], pitch 80B

    const int tid  = threadIdx.x;
    const int wave = tid >> 5;
    const int lane = tid & 31;
    const int lo   = lane & 15;
    const int hi   = lane >> 4;

    v8f acc[2] = {};
    const int t0    = wave * 2;
    const int o_t   = t0 >> 2;          // both tiles of a wave share o_t
    const int n_t0  = t0 & 3;

    for (int c0 = 0; c0 < C_; c0 += 32) {
        __syncthreads();
        // activations: float4 reads along n, transposed bf16 writes -> xs[n][k]
        // 32*16 = 512 chunks over 256 threads -> exactly 2 straight-line steps
        #pragma unroll
        for (int s = 0; s < 2; ++s) {
            const int i  = tid + s * 256;
            const int kk = i >> 4, nn4 = (i & 15) * 4;
            float4 f = *(const float4*)&X[((size_t)b * C_ + (c0 + kk)) * HW_ + n0 + nn4];
            xs[nn4 + 0][kk] = f2bf(f.x);
            xs[nn4 + 1][kk] = f2bf(f.y);
            xs[nn4 + 2][kk] = f2bf(f.z);
            xs[nn4 + 3][kk] = f2bf(f.w);
        }
        // weights: float4 reads along k, packed-pair writes -> wsm[o][k]
        #pragma unroll
        for (int s = 0; s < 2; ++s) {
            const int i  = tid + s * 256;
            const int oo = i >> 3, k4 = (i & 7) * 4;
            float4 f = *(const float4*)&W[(size_t)oo * C_ + c0 + k4];
            *(unsigned*)&wsm[oo][k4]     = pack2bf(f.x, f.y);
            *(unsigned*)&wsm[oo][k4 + 2] = pack2bf(f.z, f.w);
        }
        __syncthreads();

        v16bf a = fragA(&wsm[o_t * 16 + lo][0], hi, 0);
        #pragma unroll
        for (int i = 0; i < 2; ++i) {
            v16bf bm = fragB(&xs[(n_t0 + i) * 16 + lo][0], hi, 0);
            acc[i] = __builtin_amdgcn_wmma_f32_16x16x32_bf16(
                         false, a, false, bm, (short)0, acc[i], false, false);
        }
    }

    // Epilogue: bias (+pos for k), convert to bf16, write workspace layouts
    #pragma unroll
    for (int i = 0; i < 2; ++i) {
        const int n_t = n_t0 + i;
        #pragma unroll
        for (int r = 0; r < 8; ++r) {
            const int o_l   = o_t * 16 + r + hi * 8;   // 0..63 within slice
            const int n_abs = n0 + n_t * 16 + lo;
            const int o_abs = orow0 + o_l;
            float val = acc[i][r] + bias[o_l];
            if (pj == 0) {
                qWS[((size_t)b * HW_ + n_abs) * C8_ + o_abs] = f2bf(val);
            } else if (pj == 1) {
                const int y = n_abs >> 6, x = n_abs & 63;
                val += ht[o_l * 64 + y] + wt[o_l * 64 + x];   // fold pos into k
                kWS[((size_t)b * HW_ + n_abs) * C8_ + o_abs] = f2bf(val);
            } else {
                vWS[((size_t)b * C_ + o_abs) * HW_ + n_abs] = f2bf(val);
            }
        }
    }
}

// ---------------------------------------------------------------------------
// Kernel 2: flash-attention over m, fused PV + gamma*out + residual.
//   blockIdx.x : 64-row n-block, blockIdx.y : batch. 512 threads = 16 waves.
//   Wave w: n-tile = w&3 (shared by S and acc), c-group = w>>2 (128 channels).
//   Staging uses GLOBAL_LOAD_ASYNC_TO_LDS_B128 (ASYNCcnt).
// ---------------------------------------------------------------------------
__global__ __launch_bounds__(512) void attn_flash_kernel(
    const unsigned short* __restrict__ qWS, const unsigned short* __restrict__ kWS,
    const unsigned short* __restrict__ vWS, const float* __restrict__ fe,
    const float* __restrict__ gamma, float* __restrict__ out)
{
    const int b  = blockIdx.y;
    const int n0 = blockIdx.x * 64;
    const int tid  = threadIdx.x;
    const int wave = tid >> 5;
    const int lane = tid & 31;
    const int lo   = lane & 15;
    const int hi   = lane >> 4;
    const int n_t  = wave & 3;
    const int cg   = wave >> 2;

    __shared__ __align__(16) unsigned short qs[64][72];   // q  [n][k]
    __shared__ __align__(16) unsigned short kt[64][72];   // k'^T [m][C8]  (= Bt[N][K])
    __shared__ __align__(16) unsigned short ps[64][72];   // probs [n][m]
    __shared__ __align__(16) unsigned short vs[512][72];  // v [c][m]      (= Bt[N][K])
    __shared__ __align__(16) float Stile[64][68];
    __shared__ float row_m[64], row_l[64], row_scale[64];

    // q block: one async b128 per thread (64 rows x 8 chunks = 512)
    {
        const int nn = tid >> 3, c4 = (tid & 7) * 8;
        async_b128(&qs[nn][c4], &qWS[((size_t)b * HW_ + n0 + nn) * C8_ + c4]);
    }
    if (tid < 64) { row_m[tid] = -3.0e38f; row_l[tid] = 0.0f; }
    wait_async0();
    __syncthreads();

    // q A-fragments (K = C8 = 64 -> 2 fragments) kept in VGPRs for whole pass
    v16bf qa[2];
    qa[0] = fragA(&qs[n_t * 16 + lo][0], hi, 0);
    qa[1] = fragA(&qs[n_t * 16 + lo][0], hi, 1);

    v8f acc[8] = {};

    const int kt_mm = tid >> 3, kt_c4 = (tid & 7) * 8;

    for (int m0 = 0; m0 < HW_; m0 += 64) {
        __syncthreads();
        // k'^T tile: one async b128 per thread
        async_b128(&kt[kt_mm][kt_c4],
                   &kWS[((size_t)b * HW_ + m0 + kt_mm) * C8_ + kt_c4]);
        // v tile: 8 async b128 per thread (512 rows x 8 chunks)
        #pragma unroll
        for (int j = 0; j < 8; ++j) {
            const int i  = tid + j * 512;
            const int cc = i >> 3, c4 = (i & 7) * 8;
            async_b128(&vs[cc][c4], &vWS[((size_t)b * C_ + cc) * HW_ + m0 + c4]);
        }
        if (m0 + 64 < HW_) {                              // global_prefetch_b8
            __builtin_prefetch(&kWS[((size_t)b * HW_ + m0 + 64 + (tid >> 3)) * C8_], 0, 1);
            __builtin_prefetch(&vWS[((size_t)b * C_ + tid) * HW_ + m0 + 64], 0, 1);
        }
        wait_async0();
        __syncthreads();

        // --- S tile (16x16) = q[n_t] x k'[:, cg]  (2 x wmma, K=64) ---
        {
            v8f s = {};
            #pragma unroll
            for (int k2 = 0; k2 < 2; ++k2) {
                v16bf kb = fragB(&kt[cg * 16 + lo][0], hi, k2);
                s = __builtin_amdgcn_wmma_f32_16x16x32_bf16(
                        false, qa[k2], false, kb, (short)0, s, false, false);
            }
            #pragma unroll
            for (int r = 0; r < 8; ++r)
                Stile[n_t * 16 + r + hi * 8][cg * 16 + lo] = s[r];
        }
        __syncthreads();

        // --- online softmax: 8 lanes per row, shfl_xor reduction ---
        {
            const int row   = tid >> 3;
            const int cbase = (tid & 7) * 8;
            float4 s0 = *(const float4*)&Stile[row][cbase];
            float4 s1 = *(const float4*)&Stile[row][cbase + 4];
            float lm = fmaxf(fmaxf(fmaxf(s0.x, s0.y), fmaxf(s0.z, s0.w)),
                             fmaxf(fmaxf(s1.x, s1.y), fmaxf(s1.z, s1.w)));
            lm = fmaxf(lm, __shfl_xor(lm, 1));
            lm = fmaxf(lm, __shfl_xor(lm, 2));
            lm = fmaxf(lm, __shfl_xor(lm, 4));
            const float mOld = row_m[row];
            const float mNew = fmaxf(mOld, lm);
            float p0 = __expf(s0.x - mNew), p1 = __expf(s0.y - mNew);
            float p2 = __expf(s0.z - mNew), p3 = __expf(s0.w - mNew);
            float p4 = __expf(s1.x - mNew), p5 = __expf(s1.y - mNew);
            float p6 = __expf(s1.z - mNew), p7 = __expf(s1.w - mNew);
            unsigned* pp = (unsigned*)&ps[row][cbase];
            pp[0] = pack2bf(p0, p1);
            pp[1] = pack2bf(p2, p3);
            pp[2] = pack2bf(p4, p5);
            pp[3] = pack2bf(p6, p7);
            float ls = ((p0 + p1) + (p2 + p3)) + ((p4 + p5) + (p6 + p7));
            ls += __shfl_xor(ls, 1);
            ls += __shfl_xor(ls, 2);
            ls += __shfl_xor(ls, 4);
            if ((tid & 7) == 0) {
                float scale = __expf(mOld - mNew);
                row_scale[row] = scale;
                row_m[row]     = mNew;
                row_l[row]     = row_l[row] * scale + ls;
            }
        }
        __syncthreads();

        // --- rescale acc, then acc += p(16x64) x v^T(64x128)  (16 x wmma) ---
        float sc[8];
        #pragma unroll
        for (int r = 0; r < 8; ++r) sc[r] = row_scale[n_t * 16 + r + hi * 8];
        #pragma unroll
        for (int ct = 0; ct < 8; ++ct)
            #pragma unroll
            for (int r = 0; r < 8; ++r) acc[ct][r] *= sc[r];

        v16bf pa[2];
        pa[0] = fragA(&ps[n_t * 16 + lo][0], hi, 0);
        pa[1] = fragA(&ps[n_t * 16 + lo][0], hi, 1);
        #pragma unroll
        for (int ct = 0; ct < 8; ++ct) {
            const unsigned short* vrow = &vs[cg * 128 + ct * 16 + lo][0];
            #pragma unroll
            for (int k2 = 0; k2 < 2; ++k2) {
                v16bf vb = fragB(vrow, hi, k2);
                acc[ct] = __builtin_amdgcn_wmma_f32_16x16x32_bf16(
                              false, pa[k2], false, vb, (short)0, acc[ct], false, false);
            }
        }
    }
    __syncthreads();

    // --- epilogue: out = gamma * (acc / l) + final_encoded ---
    const float g = gamma[0];
    float invl[8];
    #pragma unroll
    for (int r = 0; r < 8; ++r) invl[r] = 1.0f / row_l[n_t * 16 + r + hi * 8];
    #pragma unroll
    for (int ct = 0; ct < 8; ++ct) {
        const int c_abs = cg * 128 + ct * 16 + lo;
        #pragma unroll
        for (int r = 0; r < 8; ++r) {
            const int n_abs = n0 + n_t * 16 + r + hi * 8;
            const size_t idx = ((size_t)b * C_ + c_abs) * HW_ + n_abs;
            out[idx] = g * (acc[ct][r] * invl[r]) + fe[idx];
        }
    }
}

// ---------------------------------------------------------------------------
extern "C" void kernel_launch(void* const* d_in, const int* in_sizes, int n_in,
                              void* d_out, int out_size, void* d_ws, size_t ws_size,
                              hipStream_t stream) {
    const float* fe    = (const float*)d_in[0];
    const float* total = (const float*)d_in[1];
    const float* Wq    = (const float*)d_in[2];
    const float* bq    = (const float*)d_in[3];
    const float* Wk    = (const float*)d_in[4];
    const float* bk    = (const float*)d_in[5];
    const float* Wv    = (const float*)d_in[6];
    const float* bv    = (const float*)d_in[7];
    const float* ht    = (const float*)d_in[8];
    const float* wt    = (const float*)d_in[9];
    const float* gamma = (const float*)d_in[10];
    float* out = (float*)d_out;

    unsigned short* qWS = (unsigned short*)d_ws;               // 2 MB
    unsigned short* kWS = qWS + (size_t)4 * HW_ * C8_;         // 2 MB
    unsigned short* vWS = kWS + (size_t)4 * HW_ * C8_;         // 16 MB

    attn_proj_kernel<<<dim3(64, 10, 4), 256, 0, stream>>>(
        fe, total, Wq, bq, Wk, bk, Wv, bv, ht, wt, qWS, kWS, vWS);
    attn_flash_kernel<<<dim3(64, 4), 512, 0, stream>>>(
        qWS, kWS, vWS, fe, gamma, out);
}
